// DeployablePPOPolicy_gat2_85555748537213
// MI455X (gfx1250) — matrix operand order repaired and checked by reading, not compiled, not run on hardware
//
#include <hip/hip_runtime.h>

typedef __attribute__((ext_vector_type(16))) _Float16 v16h;
typedef __attribute__((ext_vector_type(8)))  float    v8f;
typedef __attribute__((ext_vector_type(4)))  int      i32x4;

#if defined(__has_builtin)
#  if __has_builtin(__builtin_amdgcn_global_load_async_to_lds_b128)
#    define USE_ASYNC_LDS 1
#  endif
#  if __has_builtin(__builtin_amdgcn_s_wait_asynccnt)
#    define WAIT_ASYNC() __builtin_amdgcn_s_wait_asynccnt(0)
#  endif
#endif
#ifndef USE_ASYNC_LDS
#  define USE_ASYNC_LDS 0
#endif
#ifndef WAIT_ASYNC
#  define WAIT_ASYNC() asm volatile("s_wait_asynccnt 0x0" ::: "memory")
#endif

// ---------------------------------------------------------------------------
// CSR construction (dst is fixed across all 5 GAT layers -> build once)
// ---------------------------------------------------------------------------
__global__ void zero_i32(int* p, int n) {
    int i = blockIdx.x * blockDim.x + threadIdx.x;
    if (i < n) p[i] = 0;
}

__global__ void copy_i32(const int* a, int* b, int n) {
    int i = blockIdx.x * blockDim.x + threadIdx.x;
    if (i < n) b[i] = a[i];
}

__global__ void count_indeg(const long long* __restrict__ ei, int E, int T,
                            int* __restrict__ indeg) {
    int i = blockIdx.x * blockDim.x + threadIdx.x;
    if (i >= E + T) return;
    int dst = (i < E) ? (int)ei[E + i] : (i - E);   // self loop for i >= E
    atomicAdd(&indeg[dst], 1);
}

// Single-block exclusive scan over T<=32768 (1024 thr x 32 elems each)
__global__ void scan_rowptr(const int* __restrict__ indeg, int* __restrict__ rowptr, int T) {
    __shared__ int s[1024];
    int t = threadIdx.x;
    int base = t * 32;
    int acc = 0;
    for (int i = 0; i < 32; ++i) { int idx = base + i; acc += (idx < T) ? indeg[idx] : 0; }
    s[t] = acc;
    __syncthreads();
    for (int off = 1; off < 1024; off <<= 1) {
        int v = (t >= off) ? s[t - off] : 0;
        __syncthreads();
        s[t] += v;
        __syncthreads();
    }
    int excl = (t == 0) ? 0 : s[t - 1];
    for (int i = 0; i < 32; ++i) {
        int idx = base + i;
        if (idx < T) { rowptr[idx] = excl; excl += indeg[idx]; }
    }
    if (t == 1023) rowptr[T] = excl;
}

__global__ void fill_col(const long long* __restrict__ ei, int E, int T,
                         int* __restrict__ cursor, int* __restrict__ col) {
    int i = blockIdx.x * blockDim.x + threadIdx.x;
    if (i >= E + T) return;
    int src, dst;
    if (i < E) { src = (int)ei[i]; dst = (int)ei[E + i]; }
    else       { src = dst = i - E; }
    int pos = atomicAdd(&cursor[dst], 1);
    col[pos] = src;
}

// ---------------------------------------------------------------------------
// Fused WMMA GEMM pair: outL = act(A) @ WL^T, outR = act(A) @ WR^T
// 256 threads = 8 waves; block tile = 16 rows; A staged once to LDS as f16.
// Waves 0-3 -> WL n-tiles 0-3; waves 4-7 -> WR n-tiles 0-3.
// Fragment layouts per CDNA5 ISA 7.12.2 (wave32).
// ---------------------------------------------------------------------------
template <int KD, bool RELU_IN>
__global__ void wmma_gemm_dual(const float* __restrict__ A,
                               const float* __restrict__ WL, const float* __restrict__ WR,
                               float* __restrict__ outL, float* __restrict__ outR) {
    __shared__ _Float16 As[16 * 64];
    const int tid  = threadIdx.x;
    const int row0 = blockIdx.x * 16;

#if USE_ASYNC_LDS
    __shared__ float Araw[16 * KD];
    {
        constexpr int BYTES = 16 * KD * 4;   // tile is contiguous in global (row-major)
        if (tid * 16 < BYTES) {
            const char* g = (const char*)(A + (size_t)row0 * KD) + tid * 16;
            __builtin_amdgcn_global_load_async_to_lds_b128(
                (__attribute__((address_space(1))) i32x4*)g,
                (__attribute__((address_space(3))) i32x4*)((char*)Araw + tid * 16),
                0, 0);
        }
        WAIT_ASYNC();
        __syncthreads();
    }
    for (int i = tid; i < 16 * 64; i += 256) {
        int r = i >> 6, k = i & 63;
        float v = (k < KD) ? Araw[r * KD + (k < KD ? k : 0)] : 0.f;
        if (RELU_IN) v = fmaxf(v, 0.f);
        As[i] = (_Float16)v;
    }
#else
    for (int i = tid; i < 16 * 64; i += 256) {
        int r = i >> 6, k = i & 63;
        int kc = (k < KD) ? k : 0;                    // clamped index: no branchy loads
        float v = A[(size_t)(row0 + r) * KD + kc];
        if (RELU_IN) v = fmaxf(v, 0.f);
        v = (k < KD) ? v : 0.f;
        As[i] = (_Float16)v;
    }
#endif
    __syncthreads();

    const int wave = tid >> 5, lane = tid & 31;
    const int r  = lane & 15;
    const int hi = lane >> 4;
    const int ncol = (wave & 3) * 16 + r;
    const float* W   = (wave < 4) ? WL : WR;
    float*       out = (wave < 4) ? outL : outR;

    v8f c = {};
    const int nchunk = (KD > 32) ? 2 : 1;
    for (int ch = 0; ch < nchunk; ++ch) {
        v16h a, b;
        // A 16x32 f16: lanes<16: a[0..7]=K0..7, a[8..15]=K16..23; lanes>=16: +8
        int kbaseA = ch * 32 + hi * 8;
        for (int i = 0; i < 8; ++i) {
            a[i]     = As[r * 64 + kbaseA + i];
            a[8 + i] = As[r * 64 + kbaseA + 16 + i];
        }
        // B 32x16 f16: lanes<16: b[0..7]=K0..7, b[8..15]=K16..23 of column r; lanes>=16: +8
        int kbaseB = ch * 32 + hi * 8;
        for (int i = 0; i < 16; ++i) {
            int k  = kbaseB + (i < 8 ? i : i + 8);
            int kc = (k < KD) ? k : 0;
            float wv = W[(size_t)ncol * KD + kc];
            b[i] = (_Float16)((k < KD) ? wv : 0.f);
        }
        c = __builtin_amdgcn_wmma_f32_16x16x32_f16(false, a, false, b, (short)0, c,
                                                   false, false);
    }
    // D f32 16x16: VGPR v -> M = v + 8*hi, N = lane&15
    for (int vv = 0; vv < 8; ++vv) {
        int mrow = vv + 8 * hi;
        out[(size_t)(row0 + mrow) * 64 + ncol] = c[vv];
    }
}

// Single-output WMMA GEMM with bias (head transform t7)
template <int KD, bool RELU_IN, bool HAS_BIAS>
__global__ void wmma_gemm64(const float* __restrict__ A, const float* __restrict__ W,
                            const float* __restrict__ bias, float* __restrict__ out) {
    __shared__ _Float16 As[16 * 64];
    const int tid  = threadIdx.x;
    const int row0 = blockIdx.x * 16;

    for (int i = tid; i < 16 * 64; i += 128) {
        int r = i >> 6, k = i & 63;
        int kc = (k < KD) ? k : 0;
        float v = A[(size_t)(row0 + r) * KD + kc];
        if (RELU_IN) v = fmaxf(v, 0.f);
        v = (k < KD) ? v : 0.f;
        As[i] = (_Float16)v;
    }
    __syncthreads();

    const int wave = tid >> 5, lane = tid & 31;
    const int r  = lane & 15;
    const int hi = lane >> 4;
    const int ncol = wave * 16 + r;

    v8f c = {};
    const int nchunk = (KD > 32) ? 2 : 1;
    for (int ch = 0; ch < nchunk; ++ch) {
        v16h a, b;
        int kbaseA = ch * 32 + hi * 8;
        for (int i = 0; i < 8; ++i) {
            a[i]     = As[r * 64 + kbaseA + i];
            a[8 + i] = As[r * 64 + kbaseA + 16 + i];
        }
        int kbaseB = ch * 32 + hi * 8;
        for (int i = 0; i < 16; ++i) {
            int k  = kbaseB + (i < 8 ? i : i + 8);
            int kc = (k < KD) ? k : 0;
            float wv = W[(size_t)ncol * KD + kc];
            b[i] = (_Float16)((k < KD) ? wv : 0.f);
        }
        c = __builtin_amdgcn_wmma_f32_16x16x32_f16(false, a, false, b, (short)0, c,
                                                   false, false);
    }
    float bb = HAS_BIAS ? bias[ncol] : 0.f;
    for (int vv = 0; vv < 8; ++vv) {
        int mrow = vv + 8 * hi;
        out[(size_t)(row0 + mrow) * 64 + ncol] = c[vv] + bb;
    }
}

// ---------------------------------------------------------------------------
// GATv2 edge softmax + aggregation: one wave32 per destination node,
// ONLINE softmax (single pass over edges -> xl[src] rows read exactly once).
// Lane holds feature dims {2*lane, 2*lane+1} as float2 (coalesced b64 loads).
// ---------------------------------------------------------------------------
__global__ void gat_aggregate(const float* __restrict__ xl, const float* __restrict__ xr,
                              const float* __restrict__ att, const float* __restrict__ bias,
                              const int* __restrict__ rowptr, const int* __restrict__ col,
                              float* __restrict__ hout, int T) {
    const int wave = threadIdx.x >> 5, lane = threadIdx.x & 31;
    const int node = blockIdx.x * 4 + wave;
    if (node >= T) return;

    const float2* xl2 = (const float2*)xl;
    const float2  xrv = ((const float2*)xr)[(size_t)node * 32 + lane];
    const float2  av  = ((const float2*)att)[lane];

    const int start = rowptr[node];
    const int end   = rowptr[node + 1];

    float m = -3.4e38f, den = 0.f, acc0 = 0.f, acc1 = 0.f;
    for (int j = start; j < end; ++j) {
        int src = col[j];
        float2 xv = xl2[(size_t)src * 32 + lane];
        float s0 = xv.x + xrv.x, s1 = xv.y + xrv.y;
        s0 = s0 > 0.f ? s0 : 0.2f * s0;               // leaky_relu(., 0.2)
        s1 = s1 > 0.f ? s1 : 0.2f * s1;
        float red = s0 * av.x + s1 * av.y;
        for (int o = 16; o > 0; o >>= 1) red += __shfl_xor(red, o, 32);
        // online softmax rescale (m, den, red lane-uniform)
        float mnew  = fmaxf(m, red);
        float scale = __expf(m - mnew);
        float w     = __expf(red - mnew);
        den  = den  * scale + w;
        acc0 = acc0 * scale + w * xv.x;
        acc1 = acc1 * scale + w * xv.y;
        m = mnew;
    }
    float inv = 1.f / den;
    const float2 bv2 = ((const float2*)bias)[lane];
    float2 o;
    o.x = acc0 * inv + bv2.x;
    o.y = acc1 * inv + bv2.y;
    ((float2*)hout)[(size_t)node * 32 + lane] = o;
}

// ---------------------------------------------------------------------------
// Pooling + heads
// ---------------------------------------------------------------------------
__global__ void graph_mean(const float* __restrict__ h, float* __restrict__ mu, int Npg) {
    __shared__ float s[256];
    int g = blockIdx.x, t = threadIdx.x;
    int dim = t & 63, grp = t >> 6;
    float acc = 0.f;
    for (int n = grp; n < Npg; n += 4) acc += h[((size_t)g * Npg + n) * 64 + dim];
    s[t] = acc;
    __syncthreads();
    if (t < 64) mu[g * 64 + t] = (s[t] + s[t + 64] + s[t + 128] + s[t + 192]) / (float)Npg;
}

__global__ void g_transform(const float* __restrict__ mu, const float* __restrict__ w,
                            const float* __restrict__ b, float* __restrict__ graw) {
    int g = blockIdx.x, j = threadIdx.x;
    float acc = b[j];
    for (int k = 0; k < 64; ++k) acc += mu[g * 64 + k] * w[j * 64 + k];
    graw[g * 64 + j] = acc;
}

__global__ void head_kernel(const float* __restrict__ graw, const float* __restrict__ lbuf,
                            const int* __restrict__ reachable,
                            const float* __restrict__ t5pw, const float* __restrict__ t5pb,
                            const float* __restrict__ t5vw, const float* __restrict__ t5vb,
                            const float* __restrict__ pw, const float* __restrict__ pb,
                            float* __restrict__ out_logits, float* __restrict__ qbuf,
                            int T, int Npg) {
    int t = blockIdx.x * blockDim.x + threadIdx.x;
    if (t >= T) return;
    int g = t / Npg;
    float p = 0.f, q = 0.f;
    for (int k = 0; k < 64; ++k) {
        float gv = graw[g * 64 + k]; gv = gv > 0.f ? gv : 0.f;
        float lv = lbuf[(size_t)t * 64 + k]; lv = lv > 0.f ? lv : 0.f;
        p += gv * t5pw[k] + lv * t5pw[64 + k];
        q += gv * t5vw[k] + lv * t5vw[64 + k];
    }
    p += t5pb[0];
    q += t5vb[0];
    out_logits[t] = p * pw[0] + pb[0];
    qbuf[t] = reachable[t] ? q : -1e20f;
}

__global__ void value_reduce(const float* __restrict__ qbuf, const float* __restrict__ vw,
                             const float* __restrict__ vb, float* __restrict__ out_value,
                             int Npg) {
    __shared__ float s[256];
    int g = blockIdx.x, t = threadIdx.x;
    float m = -3.4e38f;
    for (int n = t; n < Npg; n += 256) m = fmaxf(m, qbuf[(size_t)g * Npg + n]);
    s[t] = m;
    __syncthreads();
    for (int o = 128; o > 0; o >>= 1) {
        if (t < o) s[t] = fmaxf(s[t], s[t + o]);
        __syncthreads();
    }
    if (t == 0) out_value[g] = s[0] * vw[0] + vb[0];
}

// ---------------------------------------------------------------------------
// Launch
// ---------------------------------------------------------------------------
extern "C" void kernel_launch(void* const* d_in, const int* in_sizes, int n_in,
                              void* d_out, int out_size, void* d_ws, size_t ws_size,
                              hipStream_t stream) {
    const float*     x    = (const float*)d_in[0];
    const long long* ei   = (const long long*)d_in[1];
    const int*  reachable = (const int*)d_in[2];
    const float* Wl0  = (const float*)d_in[3];
    const float* Wr0  = (const float*)d_in[4];
    const float* att0 = (const float*)d_in[5];
    const float* b0   = (const float*)d_in[6];
    const float* Wl   = (const float*)d_in[7];
    const float* Wr   = (const float*)d_in[8];
    const float* att  = (const float*)d_in[9];
    const float* bv   = (const float*)d_in[10];
    const float* t6w  = (const float*)d_in[11];
    const float* t6b  = (const float*)d_in[12];
    const float* t7w  = (const float*)d_in[13];
    const float* t7b  = (const float*)d_in[14];
    const float* t5pw = (const float*)d_in[15];
    const float* t5pb = (const float*)d_in[16];
    const float* t5vw = (const float*)d_in[17];
    const float* t5vb = (const float*)d_in[18];
    const float* pw   = (const float*)d_in[19];
    const float* pb   = (const float*)d_in[20];
    const float* vw   = (const float*)d_in[21];
    const float* vb   = (const float*)d_in[22];

    const int T   = in_sizes[0] / 8;     // F_IN = 8
    const int E   = in_sizes[1] / 2;     // edge_index is [2, E]
    const int Npg = 1024;                // n_per_graph (fixed shape)
    const int Bg  = T / Npg;

    // workspace layout
    char* w = (char*)d_ws;
    auto take = [&](size_t bytes) { char* p = w; w += (bytes + 255) & ~(size_t)255; return p; };
    float* h0     = (float*)take((size_t)T * 64 * 4);
    float* h1     = (float*)take((size_t)T * 64 * 4);
    float* xlbuf  = (float*)take((size_t)T * 64 * 4);
    float* xrbuf  = (float*)take((size_t)T * 64 * 4);
    int*   indeg  = (int*)take((size_t)T * 4);
    int*   rowptr = (int*)take((size_t)(T + 1) * 4);
    int*   cursor = (int*)take((size_t)(T + 1) * 4);
    int*   col    = (int*)take((size_t)(E + T) * 4);
    float* mu     = (float*)take((size_t)Bg * 64 * 4);
    float* graw   = (float*)take((size_t)Bg * 64 * 4);
    float* qbuf   = (float*)take((size_t)T * 4);

    float* out_logits = (float*)d_out;
    float* out_value  = (float*)d_out + T;

    const int ET = E + T;

    // --- CSR by destination (once; reused by all 5 layers) ---
    zero_i32<<<(T + 255) / 256, 256, 0, stream>>>(indeg, T);
    count_indeg<<<(ET + 255) / 256, 256, 0, stream>>>(ei, E, T, indeg);
    scan_rowptr<<<1, 1024, 0, stream>>>(indeg, rowptr, T);
    copy_i32<<<(T + 256) / 256, 256, 0, stream>>>(rowptr, cursor, T + 1);
    fill_col<<<(ET + 255) / 256, 256, 0, stream>>>(ei, E, T, cursor, col);

    const int gemmGrid = T / 16;
    const int aggGrid  = (T + 3) / 4;

    // --- layer 0: K=8 (zero-padded into one WMMA), no input relu ---
    wmma_gemm_dual<8, false><<<gemmGrid, 256, 0, stream>>>(x, Wl0, Wr0, xlbuf, xrbuf);
    gat_aggregate<<<aggGrid, 128, 0, stream>>>(xlbuf, xrbuf, att0, b0, rowptr, col, h0, T);

    // --- layers 1..4: relu on input, K=64; xl/xr fused in one GEMM pass ---
    float* hin = h0; float* hout = h1;
    for (int k = 0; k < 4; ++k) {
        const float* Wlk  = Wl + (size_t)k * 64 * 64;
        const float* Wrk  = Wr + (size_t)k * 64 * 64;
        const float* attk = att + (size_t)k * 64;
        const float* bk   = bv  + (size_t)k * 64;
        wmma_gemm_dual<64, true><<<gemmGrid, 256, 0, stream>>>(hin, Wlk, Wrk, xlbuf, xrbuf);
        gat_aggregate<<<aggGrid, 128, 0, stream>>>(xlbuf, xrbuf, attk, bk, rowptr, col, hout, T);
        float* tmp = hin; hin = hout; hout = tmp;
    }
    float* mu_raw = hin;   // after 5 layers (ends back at h0)

    // --- pooling + heads ---
    graph_mean<<<Bg, 256, 0, stream>>>(mu_raw, mu, Npg);
    g_transform<<<Bg, 64, 0, stream>>>(mu, t6w, t6b, graw);
    // l = mu_raw @ t7w.T + t7b  (reuse xlbuf)
    wmma_gemm64<64, false, true><<<gemmGrid, 128, 0, stream>>>(mu_raw, t7w, t7b, xlbuf);
    head_kernel<<<(T + 255) / 256, 256, 0, stream>>>(graw, xlbuf, reachable,
                                                     t5pw, t5pb, t5vw, t5vb, pw, pb,
                                                     out_logits, qbuf, T, Npg);
    value_reduce<<<Bg, 256, 0, stream>>>(qbuf, vw, vb, out_value, Npg);
}
